// FlashMLAttention_31086973288495
// MI455X (gfx1250) — matrix operand order repaired and checked by reading, not compiled, not run on hardware
//
#include <hip/hip_runtime.h>

// ---------------------------------------------------------------------------
// FlashMLAttention for MI455X (gfx1250): all matmuls via v_wmma_f32_16x16x32_bf16
// + Tensor Data Mover (tensor_load_to_lds) staging for bf16 A-tiles.
// B=2, S=2048, E=2048, H=16, DH=128, L=64, EB=128, KB=512
// ---------------------------------------------------------------------------

typedef unsigned short u16;
typedef unsigned int   u32;
typedef unsigned long long u64;
typedef __attribute__((ext_vector_type(16))) __bf16 v16bf;
typedef __attribute__((ext_vector_type(8)))  float  v8f;
typedef __attribute__((ext_vector_type(4)))  u32    u32x4;
typedef __attribute__((ext_vector_type(8)))  int    i32x8;
typedef __attribute__((ext_vector_type(4)))  int    i32x4;

#if defined(__has_builtin)
# if __has_builtin(__builtin_amdgcn_tensor_load_to_lds)
#  define HAVE_TDM 1
# endif
#endif
#ifndef HAVE_TDM
# define HAVE_TDM 0
#endif

union Frag { v16bf v; u32 u[8]; };

static __device__ __forceinline__ u32 f2bf(float f) {
    u32 u = __float_as_uint(f);
    return (u + 0x7FFFu + ((u >> 16) & 1u)) >> 16;   // RNE
}
static __device__ __forceinline__ u32 pack2(float lo, float hi) {
    return f2bf(lo) | (f2bf(hi) << 16);
}
static __device__ __forceinline__ v8f wmma_bf16(const Frag& a, const Frag& b, v8f c) {
    return __builtin_amdgcn_wmma_f32_16x16x32_bf16(false, a.v, false, b.v,
                                                   (short)0, c, false, false);
}
static __device__ __forceinline__ float gelu_f(float x) {
    float x3 = x * x * x;
    return 0.5f * x * (1.0f + tanhf(0.7978845608028654f * (x + 0.044715f * x3)));
}
static __device__ __forceinline__ float elu1_f(float x) {
    return x > 0.0f ? x + 1.0f : __expf(x);          // elu(x)+1
}

#if HAVE_TDM
// Issue one TDM 2D tile load (bf16 elements): tile_d0 contiguous elements per
// row, tile_d1 rows, row stride = stride_elems. Descriptor per CDNA5 ISA §8.
static __device__ __forceinline__ void tdm_load_2d_bf16(
    u32 lds_off, const void* gsrc, u32 tile_d0, u32 tile_d1,
    u32 tensor_d0, u32 tensor_d1, u32 stride_elems)
{
    const u64 ga = (u64)gsrc;
    u32x4 g0;
    g0.x = 1u;                                   // count=1, user descriptor
    g0.y = lds_off;                              // LDS byte address
    g0.z = (u32)ga;                              // global_addr[31:0]
    g0.w = ((u32)(ga >> 32) & 0x01FFFFFFu) | (2u << 30);  // addr[56:32] | type=2
    i32x8 g1;
    g1[0] = (int)(1u << 16);                     // workgroup_mask=0, data_size=1 (2B)
    g1[1] = (int)((tensor_d0 & 0xFFFFu) << 16);  // tensor_dim0[15:0]
    g1[2] = (int)((tensor_d0 >> 16) | ((tensor_d1 & 0xFFFFu) << 16));
    g1[3] = (int)((tensor_d1 >> 16) | (tile_d0 << 16));   // tile_dim0
    g1[4] = (int)(tile_d1 & 0xFFFFu);            // tile_dim1 (tile_dim2 = 0)
    g1[5] = (int)stride_elems;                   // tensor_dim0_stride[31:0]
    g1[6] = 0;                                   // stride hi, dim1_stride lo
    g1[7] = 0;
    const i32x4 z4 = {0, 0, 0, 0};
#if __clang_major__ >= 23
    const i32x8 z8 = {0, 0, 0, 0, 0, 0, 0, 0};
    __builtin_amdgcn_tensor_load_to_lds(g0, g1, z4, z4, z8, 0);
#else
    __builtin_amdgcn_tensor_load_to_lds(g0, g1, z4, z4, 0);
#endif
}
#endif

// Output modes / activations for the generic GEMM
#define MODE_F32    0   // fp32 row-major (ldc = ldw)
#define MODE_BF16   1   // bf16 row-major (ldc = ldw)
#define MODE_VT     2   // bf16 V-transposed: [B,H,DH,S]
#define MODE_LATENT 3   // bf16 latent:      [B,H,S,L], A column offset = head*DH
#define ACT_NONE 0
#define ACT_GELU 1
#define ACT_ELU1 2

// ---------------------------------------------------------------------------
// Generic tiled GEMM: C[M,N] = act(A[M,K] * W[K,N] + bias)
// Tile: 128 x NT x 32, 256 threads (8 waves), bf16 WMMA, fp32 accumulate.
// bf16 A-tiles staged via the Tensor Data Mover; fp32 A converted on stage.
// ---------------------------------------------------------------------------
template <typename TA, int MODE, int ACT, int NT>
__global__ __launch_bounds__(256) void gemm_k(
    const TA* __restrict__ A, int lda,
    const float* __restrict__ W, int ldw,
    const float* __restrict__ bias,
    void* __restrict__ outp, int K)
{
    __shared__ u16 As[128 * 32];     // A tile, row-major [row][k], bf16
    __shared__ u16 Wt[NT * 32];      // W tile transposed [n][k], bf16

    const int tid  = threadIdx.x;
    const int w    = tid >> 5;
    const int lane = tid & 31;
    const int hh   = lane >> 4;      // half-wave
    const int ln   = lane & 15;
    const int nBase = blockIdx.x * NT;
    const int mBase = blockIdx.y * 128;
    const int head  = blockIdx.z;
    const long aOff = (MODE == MODE_LATENT) ? (long)head * 128 : 0;

    constexpr int MT = (NT == 128) ? 2 : 1;
    const int wm = (NT == 128) ? (w & 3) : w;
    const int wn = (NT == 128) ? (w >> 2) : 0;

    v8f acc[MT][4];
#pragma unroll
    for (int i = 0; i < MT; ++i)
#pragma unroll
        for (int j = 0; j < 4; ++j)
#pragma unroll
            for (int e = 0; e < 8; ++e) acc[i][j][e] = 0.0f;

    for (int kt = 0; kt < K; kt += 32) {
        // ---- stage A tile ----
        if constexpr (sizeof(TA) == 2 && HAVE_TDM) {
            // bf16 A: 128x32 tile via Tensor Data Mover (one wave issues)
            if (w == 0) {
                tdm_load_2d_bf16((u32)(size_t)(&As[0]),
                                 A + (size_t)mBase * lda + aOff + kt,
                                 /*tile_d0=*/32, /*tile_d1=*/128,
                                 /*tensor_d0=*/(u32)lda, /*tensor_d1=*/128,
                                 /*stride=*/(u32)lda);
                __builtin_amdgcn_s_wait_tensorcnt(0);
            }
        } else {
            for (int i = tid; i < 128 * 8; i += 256) {
                const int r  = i >> 3;
                const int k4 = (i & 7) << 2;
                const TA* src = A + (size_t)(mBase + r) * lda + aOff + kt + k4;
                u32 p0, p1;
                if constexpr (sizeof(TA) == 4) {
                    const float4 f = *reinterpret_cast<const float4*>(src);
                    p0 = pack2(f.x, f.y);
                    p1 = pack2(f.z, f.w);
                } else {
                    const u32* s32 = reinterpret_cast<const u32*>(src);
                    p0 = s32[0];
                    p1 = s32[1];
                }
                u32* d = reinterpret_cast<u32*>(&As[r * 32 + k4]);
                d[0] = p0; d[1] = p1;
            }
        }
        // ---- stage W tile transposed (fp32 -> bf16) ----
        for (int i = tid; i < 32 * (NT / 4); i += 256) {
            const int kk = i / (NT / 4);
            const int n4 = (i % (NT / 4)) * 4;
            const float* src = W + (size_t)(kt + kk) * ldw + nBase + n4;
#pragma unroll
            for (int j = 0; j < 4; ++j)
                Wt[(n4 + j) * 32 + kk] = (u16)f2bf(src[j]);
        }
        if (kt + 32 < K) {   // L2 prefetch of next tiles
            __builtin_prefetch(A + (size_t)(mBase + (tid & 127)) * lda + aOff + kt + 32, 0, 1);
            __builtin_prefetch(W + (size_t)(kt + 32 + (tid & 31)) * ldw + nBase, 0, 1);
        }
        __syncthreads();

        // ---- fragments ----
        Frag af[MT], bf[4];
#pragma unroll
        for (int i = 0; i < MT; ++i) {
            const int mt  = (NT == 128) ? (wm + i * 4) : w;
            const int row = mt * 16 + ln;
#pragma unroll
            for (int v = 0; v < 8; ++v) {
                const int k = ((v & 3) << 1) + ((v >> 2) << 4) + (hh << 3);
                af[i].u[v] = *reinterpret_cast<const u32*>(&As[row * 32 + k]);
            }
        }
#pragma unroll
        for (int j = 0; j < 4; ++j) {
            const int n = (wn * 4 + j) * 16 + ln;
#pragma unroll
            for (int v = 0; v < 8; ++v) {
                const int k = ((v & 3) << 1) + ((v >> 2) << 4) + (hh << 3);
                bf[j].u[v] = *reinterpret_cast<const u32*>(&Wt[n * 32 + k]);
            }
        }
#pragma unroll
        for (int i = 0; i < MT; ++i)
#pragma unroll
            for (int j = 0; j < 4; ++j)
                acc[i][j] = wmma_bf16(af[i], bf[j], acc[i][j]);
        __syncthreads();
    }

    // ---- epilogue ----
#pragma unroll
    for (int i = 0; i < MT; ++i) {
        const int mt = (NT == 128) ? (wm + i * 4) : w;
#pragma unroll
        for (int j = 0; j < 4; ++j) {
            const int nt = wn * 4 + j;
#pragma unroll
            for (int v = 0; v < 8; ++v) {
                const int row = mBase + mt * 16 + v + (hh << 3);
                const int col = nBase + nt * 16 + ln;
                float val = acc[i][j][v] + bias[col];
                if (ACT == ACT_GELU) val = gelu_f(val);
                else if (ACT == ACT_ELU1) val = elu1_f(val);

                if (MODE == MODE_F32) {
                    ((float*)outp)[(size_t)row * ldw + col] = val;
                } else if (MODE == MODE_BF16) {
                    ((u16*)outp)[(size_t)row * ldw + col] = (u16)f2bf(val);
                } else if (MODE == MODE_VT) {
                    const int bb = row >> 11, s = row & 2047;     // S = 2048
                    const int h2 = col >> 7, dh = col & 127;      // DH = 128
                    ((u16*)outp)[(((size_t)(bb * 16 + h2) * 128) + dh) * 2048 + s] =
                        (u16)f2bf(val);
                } else {  // MODE_LATENT, [B,H,S,L]
                    const int bb = row >> 11, s = row & 2047;
                    ((u16*)outp)[(((size_t)(bb * 16 + head)) * 2048 + s) * 64 + col] =
                        (u16)f2bf(val);
                }
            }
        }
    }
}

// ---------------------------------------------------------------------------
// Block attention: per (b,h,qblock=128 rows). scores = q_lat @ k_lat^T / sqrt(L),
// softmax per 512-wide kv block (online, 128-col chunks), out = sum_blocks P@V.
// Each wave owns 16 query rows. K-latent and V^T fragments streamed from L2.
// ---------------------------------------------------------------------------
__global__ __launch_bounds__(256) void attn_k(
    const u16* __restrict__ qlat,   // [B,H,S,L] bf16
    const u16* __restrict__ klat,   // [B,H,S,L] bf16
    const u16* __restrict__ vt,     // [B,H,DH,S] bf16
    float* __restrict__ outp)       // [B,S,E] fp32
{
    __shared__ u16 Pl[8][16 * 128];  // per-wave probs scratch (32 KB)

    const int tid  = threadIdx.x;
    const int w    = tid >> 5;
    const int lane = tid & 31;
    const int hh   = lane >> 4;
    const int ln   = lane & 15;
    const int qb = blockIdx.x, h = blockIdx.y, b = blockIdx.z;
    const size_t bh = (size_t)(b * 16 + h);
    const u16* qbase = qlat + bh * 2048 * 64;
    const u16* kbase = klat + bh * 2048 * 64;
    const u16* vbase = vt   + bh * 128 * 2048;
    const int row0 = qb * 128 + w * 16;

    // Q A-fragments for L=64 (two K=32 steps), held in registers throughout
    Frag aq[2];
    {
        const u16* qr = qbase + (size_t)(row0 + ln) * 64;
#pragma unroll
        for (int ks = 0; ks < 2; ++ks)
#pragma unroll
            for (int v = 0; v < 8; ++v) {
                const int k = ks * 32 + ((v & 3) << 1) + ((v >> 2) << 4) + (hh << 3);
                aq[ks].u[v] = *reinterpret_cast<const u32*>(qr + k);
            }
    }

    v8f O[8];
#pragma unroll
    for (int t = 0; t < 8; ++t)
#pragma unroll
        for (int e = 0; e < 8; ++e) O[t][e] = 0.0f;

    const float scale = 0.125f;       // 1/sqrt(L)

    for (int kb = 0; kb < 4; ++kb) {  // kv blocks of 512
        float m8[8], d8[8];
        v8f Ob[8];
#pragma unroll
        for (int v = 0; v < 8; ++v) { m8[v] = -1e30f; d8[v] = 0.0f; }
#pragma unroll
        for (int t = 0; t < 8; ++t)
#pragma unroll
            for (int e = 0; e < 8; ++e) Ob[t][e] = 0.0f;

        for (int c = 0; c < 4; ++c) {  // 128-column chunks
            // ---- scores for 8 tiles of 16 columns ----
            v8f sc[8];
#pragma unroll
            for (int j = 0; j < 8; ++j) {
                v8f s;
#pragma unroll
                for (int e = 0; e < 8; ++e) s[e] = 0.0f;
                const int kv = kb * 512 + c * 128 + j * 16 + ln;
                const u16* kr = kbase + (size_t)kv * 64;
#pragma unroll
                for (int ks = 0; ks < 2; ++ks) {
                    Frag bk;
#pragma unroll
                    for (int v = 0; v < 8; ++v) {
                        const int k = ks * 32 + ((v & 3) << 1) + ((v >> 2) << 4) + (hh << 3);
                        bk.u[v] = *reinterpret_cast<const u32*>(kr + k);
                    }
                    s = wmma_bf16(aq[ks], bk, s);
                }
                sc[j] = s;
            }

            // ---- online softmax update (per row = per acc element v) ----
            float nm[8], alpha[8];
#pragma unroll
            for (int v = 0; v < 8; ++v) {
                float mv = m8[v];
#pragma unroll
                for (int j = 0; j < 8; ++j) mv = fmaxf(mv, sc[j][v]);
                mv = fmaxf(mv, __shfl_xor(mv, 1));
                mv = fmaxf(mv, __shfl_xor(mv, 2));
                mv = fmaxf(mv, __shfl_xor(mv, 4));
                mv = fmaxf(mv, __shfl_xor(mv, 8));
                nm[v] = mv;
                alpha[v] = __expf((m8[v] - mv) * scale);
                m8[v] = mv;
            }
#pragma unroll
            for (int v = 0; v < 8; ++v) {
                float sum = 0.0f;
#pragma unroll
                for (int j = 0; j < 8; ++j) {
                    const float p = __expf((sc[j][v] - nm[v]) * scale);
                    sc[j][v] = p;
                    sum += p;
                }
                sum += __shfl_xor(sum, 1);
                sum += __shfl_xor(sum, 2);
                sum += __shfl_xor(sum, 4);
                sum += __shfl_xor(sum, 8);
                d8[v] = d8[v] * alpha[v] + sum;
            }
#pragma unroll
            for (int t = 0; t < 8; ++t)
#pragma unroll
                for (int v = 0; v < 8; ++v) Ob[t][v] *= alpha[v];

            // ---- probs C-layout -> A-layout via per-wave LDS scratch ----
            u16* pl = &Pl[w][0];
#pragma unroll
            for (int j = 0; j < 8; ++j)
#pragma unroll
                for (int v = 0; v < 8; ++v)
                    pl[(v + (hh << 3)) * 128 + j * 16 + ln] = (u16)f2bf(sc[j][v]);

            // ---- P @ V over this chunk (4 K=32 steps, 8 DH tiles) ----
#pragma unroll
            for (int ks2 = 0; ks2 < 4; ++ks2) {
                Frag ap;
#pragma unroll
                for (int v = 0; v < 8; ++v) {
                    const int k = ks2 * 32 + (hh << 4) + (v << 1);
                    ap.u[v] = *reinterpret_cast<const u32*>(&pl[ln * 128 + k]);
                }
                const int kvb = kb * 512 + c * 128 + ks2 * 32 + (hh << 4);
#pragma unroll
                for (int t = 0; t < 8; ++t) {
                    Frag bv;
                    const u16* vr = vbase + (size_t)(t * 16 + ln) * 2048 + kvb;
#pragma unroll
                    for (int v = 0; v < 8; ++v)
                        bv.u[v] = *reinterpret_cast<const u32*>(vr + 2 * v);
                    Ob[t] = wmma_bf16(ap, bv, Ob[t]);
                }
            }
        }

        // normalize this kv block's contribution and accumulate
#pragma unroll
        for (int t = 0; t < 8; ++t)
#pragma unroll
            for (int v = 0; v < 8; ++v) O[t][v] += Ob[t][v] / d8[v];
    }

    // ---- write attention output as [B,S,E] fp32 ----
#pragma unroll
    for (int t = 0; t < 8; ++t)
#pragma unroll
        for (int v = 0; v < 8; ++v) {
            const int s = row0 + v + (hh << 3);
            const int e = h * 128 + t * 16 + ln;
            outp[((size_t)b * 2048 + s) * 2048 + e] = O[t][v];
        }
}

// ---------------------------------------------------------------------------
// Host launcher
// ---------------------------------------------------------------------------
extern "C" void kernel_launch(void* const* d_in, const int* in_sizes, int n_in,
                              void* d_out, int out_size, void* d_ws, size_t ws_size,
                              hipStream_t stream) {
    const float* x   = (const float*)d_in[0];
    const float* Wq  = (const float*)d_in[1];
    const float* bq  = (const float*)d_in[2];
    const float* Wk  = (const float*)d_in[3];
    const float* bk  = (const float*)d_in[4];
    const float* Wv  = (const float*)d_in[5];
    const float* bv  = (const float*)d_in[6];
    const float* Wo  = (const float*)d_in[7];
    const float* bo  = (const float*)d_in[8];
    const float* Wql = (const float*)d_in[9];
    const float* bql = (const float*)d_in[10];
    const float* Wkl = (const float*)d_in[11];
    const float* bkl = (const float*)d_in[12];
    const float* Wm  = (const float*)d_in[13];
    const float* bm  = (const float*)d_in[14];

    char* wsb = (char*)d_ws;
    // peak workspace: 64 MB
    u16*   wsQ     = (u16*)(wsb + 0);                       // 16 MB  bf16 [4096,2048]
    u16*   wsK     = (u16*)(wsb + (16u << 20));             // 16 MB
    u16*   wsVt    = (u16*)(wsb + (32u << 20));             // 16 MB  bf16 [B,H,DH,S]
    u16*   wsQL    = (u16*)(wsb + (48u << 20));             //  8 MB  bf16 [B,H,S,L]
    u16*   wsKL    = (u16*)(wsb + (56u << 20));             //  8 MB
    float* wsAttn  = (float*)(wsb + 0);                     // 32 MB  fp32 (reuses Q+K)
    u16*   wsMixed = (u16*)(wsb + (32u << 20));             // 16 MB  bf16 (reuses Vt)

    const dim3 blk(256);

    // Q/K projections -> bf16
    gemm_k<float, MODE_BF16, ACT_NONE, 128><<<dim3(16, 32, 1), blk, 0, stream>>>(
        x, 2048, Wq, 2048, bq, wsQ, 2048);
    gemm_k<float, MODE_BF16, ACT_NONE, 128><<<dim3(16, 32, 1), blk, 0, stream>>>(
        x, 2048, Wk, 2048, bk, wsK, 2048);
    // V projection -> transposed bf16 [B,H,DH,S]
    gemm_k<float, MODE_VT, ACT_NONE, 128><<<dim3(16, 32, 1), blk, 0, stream>>>(
        x, 2048, Wv, 2048, bv, wsVt, 2048);
    // latent projections: elu+1, batched over heads (grid.z = H); A staged by TDM
    gemm_k<u16, MODE_LATENT, ACT_ELU1, 64><<<dim3(1, 32, 16), blk, 0, stream>>>(
        wsQ, 2048, Wql, 64, bql, wsQL, 128);
    gemm_k<u16, MODE_LATENT, ACT_ELU1, 64><<<dim3(1, 32, 16), blk, 0, stream>>>(
        wsK, 2048, Wkl, 64, bkl, wsKL, 128);
    // block attention (nq=16, H=16, B=2)
    attn_k<<<dim3(16, 16, 2), blk, 0, stream>>>(wsQL, wsKL, wsVt, wsAttn);
    // latent mixer: gelu(attn @ Wm + bm), viewed as [65536,128]x[128,128]
    gemm_k<float, MODE_BF16, ACT_GELU, 128><<<dim3(1, 512, 1), blk, 0, stream>>>(
        wsAttn, 128, Wm, 128, bm, wsMixed, 128);
    // output projection -> fp32 d_out; A staged by TDM
    gemm_k<u16, MODE_F32, ACT_NONE, 128><<<dim3(16, 32, 1), blk, 0, stream>>>(
        wsMixed, 2048, Wo, 2048, bo, d_out, 2048);
}
